// KDEformer_57140244906017
// MI455X (gfx1250) — matrix operand order repaired and checked by reading, not compiled, not run on hardware
//
#include <hip/hip_runtime.h>
#include <math.h>

#define BQ 2
#define HQ 8
#define NQ 8192
#define DQ 64
#define BHN (BQ*HQ)
#define NP 7
#define BUCKET 64
#define NB (NQ/BUCKET)      // 128 buckets per head
#define SS 256              // residual samples
#define DP 65               // D+1
#define DPAD 80             // padded to 5x16 WMMA col tiles
#define SCALE_QK 0.35355339059327373f  // D^{-1/4}

typedef __attribute__((ext_vector_type(16))) __bf16 v16bf;
typedef __attribute__((ext_vector_type(8)))  float  v8f;

#if defined(__has_builtin)
# if __has_builtin(__builtin_amdgcn_global_load_async_to_lds_b128) && \
     __has_builtin(__builtin_amdgcn_s_wait_asynccnt)
#  define HAVE_ASYNC_LDS 1
# endif
#endif
#ifndef HAVE_ASYNC_LDS
# define HAVE_ASYNC_LDS 0
#endif

#if HAVE_ASYNC_LDS
typedef int v4i __attribute__((vector_size(16)));
typedef __attribute__((address_space(1))) v4i v4i_g;   // global
typedef __attribute__((address_space(3))) v4i v4i_l;   // LDS
// one wave instruction moves 32 lanes x 16B; tracked by ASYNCcnt
__device__ __forceinline__ void async_cp16(const __bf16* g, __bf16* l) {
  __builtin_amdgcn_global_load_async_to_lds_b128((v4i_g*)(__bf16*)g, (v4i_l*)l, 0, 0);
}
__device__ __forceinline__ void async_wait0() { __builtin_amdgcn_s_wait_asynccnt(0); }
#endif

__device__ __forceinline__ v8f wmma_bf16(v16bf a, v16bf b, v8f c) {
  return __builtin_amdgcn_wmma_f32_16x16x32_bf16(false, a, false, b, (short)0, c, false, false);
}
// A-matrix (16x32 bf16) per-lane element -> K index (interleaved halves per ISA layout)
__device__ __forceinline__ int a_kmap(int j, int hi) { return (j & 7) + ((j & 8) << 1) + (hi << 3); }
// B-matrix (32x16 bf16): same interleaved K order as A
__device__ __forceinline__ int b_kmap(int j, int hi) { return (j & 7) + ((j & 8) << 1) + (hi << 3); }

// ---- K1: LSH Gray codes, v_aug row norms, and bf16 pre-converted copies ----
__global__ __launch_bounds__(256) void k_hash(const float* __restrict__ q,
                                              const float* __restrict__ k,
                                              const float* __restrict__ v,
                                              const float* __restrict__ proj,
                                              unsigned* __restrict__ codesQ,
                                              unsigned* __restrict__ codesK,
                                              float* __restrict__ rownorm,
                                              __bf16* __restrict__ qbf,
                                              __bf16* __restrict__ kbf,
                                              __bf16* __restrict__ wbf) {
  __shared__ float pr[DQ * NP];
  int tid = threadIdx.x;
  for (int i = tid; i < DQ * NP; i += 256) pr[i] = proj[i];
  __syncthreads();
  long t = (long)blockIdx.x * 256 + tid;          // flat (bh, n)
  const float* qr = q + t * DQ;
  const float* kr = k + t * DQ;
  const float* vr = v + t * DQ;
  __bf16* qo = qbf + t * DQ;
  __bf16* ko = kbf + t * DQ;
  __bf16* wo = wbf + t * DPAD;
  float aq[NP], ak[NP];
  for (int r = 0; r < NP; ++r) { aq[r] = 0.f; ak[r] = 0.f; }
  float vs = 0.f;
  for (int d = 0; d < DQ; ++d) {
    float qd = qr[d], kd = kr[d], vd = vr[d];
    vs += vd * vd;
    qo[d] = (__bf16)(qd * SCALE_QK);
    ko[d] = (__bf16)(kd * SCALE_QK);
    wo[d] = (__bf16)vd;
    #pragma unroll
    for (int r = 0; r < NP; ++r) {
      float p = pr[d * NP + r];
      aq[r] += qd * p; ak[r] += kd * p;
    }
  }
  wo[DQ] = (__bf16)1.0f;
  for (int d = DQ + 1; d < DPAD; ++d) wo[d] = (__bf16)0.0f;
  unsigned bq = 0, bk = 0;
  #pragma unroll
  for (int r = 0; r < NP; ++r) {
    bq |= (unsigned)(aq[r] > 0.f) << r;
    bk |= (unsigned)(ak[r] > 0.f) << r;
  }
  codesQ[t] = bq ^ (bq >> 1);                     // binary-reflected Gray code == hamming perm
  codesK[t] = bk ^ (bk >> 1);
  rownorm[t] = sqrtf(vs + 1.0f);                  // ||[v,1]||
}

// ---- K2: in-LDS bitonic argsort, stable via packed index ----
__global__ __launch_bounds__(1024) void k_sort(const unsigned* __restrict__ codesK,
                                               const unsigned* __restrict__ codesQ,
                                               int* __restrict__ Kidx,
                                               int* __restrict__ Qidx,
                                               int* __restrict__ invQ) {
  __shared__ unsigned keys[NQ];                   // 32KB
  int bh = blockIdx.x >> 1;
  int which = blockIdx.x & 1;                     // 0 = key codes, 1 = query codes
  const unsigned* codes = which ? codesQ : codesK;
  long base = (long)bh * NQ;
  for (int i = threadIdx.x; i < NQ; i += 1024)
    keys[i] = (codes[base + i] << 13) | (unsigned)i;
  for (unsigned kk = 2; kk <= NQ; kk <<= 1) {
    for (unsigned j = kk >> 1; j > 0; j >>= 1) {
      __syncthreads();
      for (unsigned i = threadIdx.x; i < NQ; i += 1024) {
        unsigned l = i ^ j;
        if (l > i) {
          unsigned a = keys[i], b = keys[l];
          bool up = ((i & kk) == 0);
          if ((a > b) == up) { keys[i] = b; keys[l] = a; }
        }
      }
    }
  }
  __syncthreads();
  for (int p = threadIdx.x; p < NQ; p += 1024) {
    int idx = (int)(keys[p] & (NQ - 1));
    if (which) { Qidx[base + p] = idx; invQ[base + idx] = p; }
    else       { Kidx[base + p] = idx; }
  }
}

// ---- K3: block-sparse attention (64x64 bucket, WMMA bf16, async gathers) ----
__global__ __launch_bounds__(128) void k_block_attn(const __bf16* __restrict__ qbf,
                                                    const __bf16* __restrict__ kbf,
                                                    const __bf16* __restrict__ wbf,
                                                    const int* __restrict__ Qidx,
                                                    const int* __restrict__ Kidx,
                                                    float* __restrict__ res) {
  __shared__ __bf16 qs[64][DQ];     // 8KB
  __shared__ __bf16 ks[64][DQ];     // 8KB
  __shared__ __bf16 ws[64][DPAD];   // 10KB (v_aug, zero-padded)
  __shared__ __bf16 se[64][64];     // 8KB exp(scores)
  __shared__ int qtok[64], ktok[64];
  int g = blockIdx.x;
  int bh = g / NB, bucket = g % NB;
  long sbase = (long)bh * NQ + bucket * 64;       // sorted-order row base
  int tid = threadIdx.x;
  long rbase = (long)bh * NQ;
  if (tid < 64) { qtok[tid] = Qidx[sbase + tid]; ktok[tid] = Kidx[sbase + tid]; }
  __syncthreads();
#if HAVE_ASYNC_LDS
  for (int e = tid; e < 64 * 8; e += 128) {       // 8 x 16B chunks per 64-elem row
    int r = e >> 3, c = (e & 7) * 8;
    async_cp16(qbf + (rbase + qtok[r]) * DQ + c, &qs[r][c]);
    async_cp16(kbf + (rbase + ktok[r]) * DQ + c, &ks[r][c]);
  }
  for (int e = tid; e < 64 * 10; e += 128) {      // 10 x 16B chunks per 80-elem row
    int r = e / 10, c = (e % 10) * 8;
    async_cp16(wbf + (rbase + ktok[r]) * DPAD + c, &ws[r][c]);
  }
  async_wait0();
#else
  for (int e = tid; e < 64 * DQ; e += 128) {
    int r = e >> 6, c = e & 63;
    qs[r][c] = qbf[(rbase + qtok[r]) * DQ + c];
    ks[r][c] = kbf[(rbase + ktok[r]) * DQ + c];
  }
  for (int e = tid; e < 64 * DPAD; e += 128) {
    int r = e / DPAD, c = e % DPAD;
    ws[r][c] = wbf[(rbase + ktok[r]) * DPAD + c];
  }
#endif
  __syncthreads();
  int wv = tid >> 5, lane = tid & 31, m = lane & 15, hi = lane >> 4;
  int m0 = wv * 16;
  // scores: exp(q . k^T)
  for (int ct = 0; ct < 4; ++ct) {
    v8f acc = {};
    for (int kb = 0; kb < DQ; kb += 32) {
      v16bf A, Bm;
      #pragma unroll
      for (int j = 0; j < 16; ++j) {
        A[j]  = qs[m0 + m][kb + a_kmap(j, hi)];
        Bm[j] = ks[ct * 16 + m][kb + b_kmap(j, hi)];
      }
      acc = wmma_bf16(A, Bm, acc);
    }
    #pragma unroll
    for (int e = 0; e < 8; ++e) {
      int r = m0 + e + hi * 8;
      se[r][ct * 16 + m] = (__bf16)__expf(acc[e]);
    }
  }
  __syncthreads();
  // out = exp(S) @ w_aug  (64 x 65)
  for (int t = 0; t < 5; ++t) {
    v8f acc = {};
    for (int kb = 0; kb < 64; kb += 32) {
      v16bf A, Bm;
      #pragma unroll
      for (int j = 0; j < 16; ++j) {
        A[j]  = se[m0 + m][kb + a_kmap(j, hi)];
        Bm[j] = ws[kb + b_kmap(j, hi)][t * 16 + m];
      }
      acc = wmma_bf16(A, Bm, acc);
    }
    #pragma unroll
    for (int e = 0; e < 8; ++e) {
      int r = m0 + e + hi * 8;
      int c = t * 16 + m;
      if (c < DP) res[(sbase + r) * DP + c] = acc[e];
    }
  }
}

// ---- K4: Gram = ws^T ws via WMMA, async K-chunk staging from wbf ----
__global__ __launch_bounds__(256) void k_gram(const __bf16* __restrict__ wbf,
                                              float* __restrict__ gram) {
  __shared__ __bf16 cw[32][DPAD];   // 5KB K-chunk of v_aug
  int bh = blockIdx.x, tid = threadIdx.x;
  int wv = tid >> 5, lane = tid & 31, m = lane & 15, hi = lane >> 4;
  v8f acc[4] = {};
  for (int n0 = 0; n0 < NQ; n0 += 32) {
    const __bf16* g = wbf + ((long)bh * NQ + n0) * DPAD;
#if HAVE_ASYNC_LDS
    for (int e = tid; e < (32 * DPAD) / 8; e += 256)
      async_cp16(g + e * 8, &cw[0][0] + e * 8);
    async_wait0();
#else
    for (int e = tid; e < 32 * DPAD; e += 256) (&cw[0][0])[e] = g[e];
#endif
    __syncthreads();
    for (int i = 0; i < 4; ++i) {
      int tt = wv + 8 * i;
      if (tt < 25) {
        int mt = tt / 5, nt = tt % 5;
        v16bf A, Bm;
        #pragma unroll
        for (int j = 0; j < 16; ++j) {
          A[j]  = cw[a_kmap(j, hi)][mt * 16 + m];   // (ws^T) tile: transposed read
          Bm[j] = cw[b_kmap(j, hi)][nt * 16 + m];
        }
        acc[i] = wmma_bf16(A, Bm, acc[i]);
      }
    }
    __syncthreads();
  }
  for (int i = 0; i < 4; ++i) {
    int tt = wv + 8 * i;
    if (tt < 25) {
      int mt = tt / 5, nt = tt % 5;
      #pragma unroll
      for (int e = 0; e < 8; ++e)
        gram[(long)bh * (DPAD * DPAD) + (mt * 16 + e + hi * 8) * DPAD + nt * 16 + m] = acc[i][e];
    }
  }
}

// ---- K5: power iteration for spectral norm ----
__global__ __launch_bounds__(128) void k_power(const float* __restrict__ gram,
                                               float* __restrict__ vnorm) {
  __shared__ float A[DP * DP];
  __shared__ float x[DP], y[DP];
  __shared__ float nrm;
  int bh = blockIdx.x, tid = threadIdx.x;
  for (int e = tid; e < DP * DP; e += 128) {
    int r = e / DP, c = e % DP;
    A[e] = gram[(long)bh * (DPAD * DPAD) + r * DPAD + c];
  }
  if (tid < DP) {
    unsigned h = (unsigned)(bh * DP + tid) * 2654435761u + 0x9E3779B9u;
    h ^= h >> 16; h *= 0x85EBCA6Bu; h ^= h >> 13;
    x[tid] = ((h >> 8) * (1.0f / 16777216.0f)) * 2.0f - 1.0f;
  }
  __syncthreads();
  if (tid == 0) {
    float s = 0.f;
    for (int i = 0; i < DP; ++i) s += x[i] * x[i];
    nrm = sqrtf(s);
  }
  __syncthreads();
  if (tid < DP) x[tid] /= nrm;
  for (int it = 0; it < 32; ++it) {
    __syncthreads();
    if (tid < DP) {
      float s = 0.f;
      for (int i = 0; i < DP; ++i) s += A[tid * DP + i] * x[i];
      y[tid] = s;
    }
    __syncthreads();
    if (tid == 0) {
      float s = 0.f;
      for (int i = 0; i < DP; ++i) s += y[i] * y[i];
      nrm = sqrtf(s);
    }
    __syncthreads();
    if (tid < DP) x[tid] = y[tid] / nrm;
  }
  if (tid == 0) vnorm[bh] = nrm;
}

// ---- K6: probabilities, CDF, categorical sampling, gather bf16 ----
__global__ __launch_bounds__(256) void k_sample(const float* __restrict__ rownorm,
                                                const float* __restrict__ vnorm,
                                                const int* __restrict__ Kidx,
                                                const float* __restrict__ k,
                                                const float* __restrict__ v,
                                                int* __restrict__ samples,
                                                __bf16* __restrict__ KpiBF,
                                                __bf16* __restrict__ WpiBF) {
  __shared__ float cdf[NQ];        // 32KB
  __shared__ float csum[256];
  __shared__ float coff[256];
  int bh = blockIdx.x, tid = threadIdx.x;
  float vn = vnorm[bh];
  long base = (long)bh * NQ;
  int c0 = tid * 32;
  float run = 0.f;
  for (int i = 0; i < 32; ++i) {
    int n = c0 + i;
    float p = rownorm[base + Kidx[base + n]] / vn + (1.0f / NQ);
    run += p;
    cdf[n] = run;
  }
  csum[tid] = run;
  __syncthreads();
  if (tid == 0) {
    float s = 0.f;
    for (int i = 0; i < 256; ++i) { coff[i] = s; s += csum[i]; }
  }
  __syncthreads();
  float off = coff[tid];
  for (int i = 0; i < 32; ++i) cdf[c0 + i] += off;
  __syncthreads();
  float total = cdf[NQ - 1];
  // one sample per thread
  unsigned h = ((unsigned)bh * 0x9E3779B9u) ^ ((unsigned)tid * 0x85EBCA6Bu) ^ 0xC2B2AE35u;
  h ^= h >> 15; h *= 0x2C1B3C6Du; h ^= h >> 12; h *= 0x297A2D39u; h ^= h >> 15;
  float u = (h >> 8) * (1.0f / 16777216.0f) * total;
  int lo = 0, hi2 = NQ - 1;
  while (lo < hi2) { int mid = (lo + hi2) >> 1; if (cdf[mid] < u) lo = mid + 1; else hi2 = mid; }
  int idx = lo;
  float pi = cdf[idx] - (idx ? cdf[idx - 1] : 0.0f);
  float sig = total / (pi * (float)SS);          // 1 / (Pnorm * S)
  samples[bh * SS + tid] = idx;
  int token = Kidx[base + idx];
  const float* kp = k + (base + token) * DQ;
  const float* vp = v + (base + token) * DQ;
  __bf16* kd = KpiBF + ((long)bh * SS + tid) * DQ;
  __bf16* wd = WpiBF + ((long)bh * SS + tid) * DPAD;
  for (int d = 0; d < DQ; ++d) {
    kd[d] = (__bf16)(kp[d] * SCALE_QK);
    wd[d] = (__bf16)(vp[d] * sig);               // sig folded into Vpi
  }
  wd[DQ] = (__bf16)sig;                          // ones column * sig
  for (int d = DQ + 1; d < DPAD; ++d) wd[d] = (__bf16)0.0f;
}

// ---- K7: importance-sampled residual, WMMA + async staging, RMW into sorted buffer --
__global__ __launch_bounds__(128) void k_residual(const __bf16* __restrict__ qbf,
                                                  const int* __restrict__ invQ,
                                                  const int* __restrict__ samples,
                                                  const __bf16* __restrict__ KpiBF,
                                                  const __bf16* __restrict__ WpiBF,
                                                  float* __restrict__ res) {
  __shared__ __bf16 qs[64][DQ];     // 8KB
  __shared__ __bf16 kp[128][DQ];    // 16KB (half of the S=256 samples per chunk)
  __shared__ __bf16 wp[128][DPAD];  // 20KB
  __shared__ __bf16 se[64][128];    // 16KB
  __shared__ int iq[64];
  __shared__ int sb[128];
  int g = blockIdx.x;
  int bh = g / (NQ / 64), blk = g % (NQ / 64);
  int n0 = blk * 64;
  int tid = threadIdx.x;
  long base = (long)bh * NQ;
  const __bf16* gq = qbf + (base + n0) * DQ;
#if HAVE_ASYNC_LDS
  for (int e = tid; e < (64 * DQ) / 8; e += 128)
    async_cp16(gq + e * 8, &qs[0][0] + e * 8);
#else
  for (int e = tid; e < 64 * DQ; e += 128) (&qs[0][0])[e] = gq[e];
#endif
  if (tid < 64) iq[tid] = invQ[base + n0 + tid];
  int wv = tid >> 5, lane = tid & 31, m = lane & 15, hi = lane >> 4;
  int m0 = wv * 16;
  v8f acc_out[5] = {};
  for (int ch = 0; ch < 2; ++ch) {
    __syncthreads();                 // previous chunk's consumers done
    int sBase = ch * 128;
    const __bf16* gk = KpiBF + ((long)bh * SS + sBase) * DQ;
    const __bf16* gw = WpiBF + ((long)bh * SS + sBase) * DPAD;
#if HAVE_ASYNC_LDS
    for (int e = tid; e < (128 * DQ) / 8; e += 128)
      async_cp16(gk + e * 8, &kp[0][0] + e * 8);
    for (int e = tid; e < (128 * DPAD) / 8; e += 128)
      async_cp16(gw + e * 8, &wp[0][0] + e * 8);
    sb[tid] = samples[bh * SS + sBase + tid] >> 6;
    async_wait0();                   // also covers the qs loads issued above
#else
    for (int e = tid; e < 128 * DQ; e += 128) (&kp[0][0])[e] = gk[e];
    for (int e = tid; e < 128 * DPAD; e += 128) (&wp[0][0])[e] = gw[e];
    sb[tid] = samples[bh * SS + sBase + tid] >> 6;
#endif
    __syncthreads();
    // scores for this sample chunk
    for (int ct = 0; ct < 8; ++ct) {
      v8f acc = {};
      for (int kb = 0; kb < DQ; kb += 32) {
        v16bf A, Bm;
        #pragma unroll
        for (int j = 0; j < 16; ++j) {
          A[j]  = qs[m0 + m][kb + a_kmap(j, hi)];
          Bm[j] = kp[ct * 16 + m][kb + b_kmap(j, hi)];
        }
        acc = wmma_bf16(A, Bm, acc);
      }
      int sc = ct * 16 + m;
      int sblk = sb[sc];
      #pragma unroll
      for (int e = 0; e < 8; ++e) {
        int r = m0 + e + hi * 8;
        float val = __expf(acc[e]);
        if ((iq[r] >> 6) == sblk) val = 0.0f;    // already covered by block attention
        se[r][sc] = (__bf16)val;
      }
    }
    __syncthreads();
    // accumulate output over this chunk's K=128
    for (int t = 0; t < 5; ++t) {
      for (int kb = 0; kb < 128; kb += 32) {
        v16bf A, Bm;
        #pragma unroll
        for (int j = 0; j < 16; ++j) {
          A[j]  = se[m0 + m][kb + a_kmap(j, hi)];
          Bm[j] = wp[kb + b_kmap(j, hi)][t * 16 + m];
        }
        acc_out[t] = wmma_bf16(A, Bm, acc_out[t]);
      }
    }
  }
  // unique-writer read-modify-write into sorted result rows
  for (int t = 0; t < 5; ++t) {
    #pragma unroll
    for (int e = 0; e < 8; ++e) {
      int r = m0 + e + hi * 8;
      int c = t * 16 + m;
      if (c < DP) {
        long o = (base + iq[r]) * DP + c;
        res[o] += acc_out[t][e];
      }
    }
  }
}

// ---- K8: normalize and unsort to original query order ----
__global__ __launch_bounds__(256) void k_final(const float* __restrict__ res,
                                               const int* __restrict__ Qidx,
                                               float* __restrict__ out) {
  long t = (long)blockIdx.x * 256 + threadIdx.x;  // (bh, sorted p)
  int bh = (int)(t / NQ), p = (int)(t % NQ);
  long base = (long)bh * NQ;
  const float* rr = res + (base + p) * DP;
  float inv = 1.0f / rr[DQ];
  int n = Qidx[base + p];
  float* op = out + (base + n) * DQ;
  for (int d = 0; d < DQ; ++d) op[d] = rr[d] * inv;
}

extern "C" void kernel_launch(void* const* d_in, const int* in_sizes, int n_in,
                              void* d_out, int out_size, void* d_ws, size_t ws_size,
                              hipStream_t stream) {
  const float* q    = (const float*)d_in[0];
  const float* k    = (const float*)d_in[1];
  const float* v    = (const float*)d_in[2];
  const float* proj = (const float*)d_in[3];
  float* out = (float*)d_out;

  char* ws = (char*)d_ws;
  size_t off = 0;
  auto alloc = [&](size_t bytes) -> void* {
    void* p = ws + off;
    off += (bytes + 255) & ~(size_t)255;
    return p;
  };
  unsigned* codesQ = (unsigned*)alloc((size_t)BHN * NQ * 4);
  unsigned* codesK = (unsigned*)alloc((size_t)BHN * NQ * 4);
  int* Kidx        = (int*)alloc((size_t)BHN * NQ * 4);
  int* Qidx        = (int*)alloc((size_t)BHN * NQ * 4);
  int* invQ        = (int*)alloc((size_t)BHN * NQ * 4);
  float* rownorm   = (float*)alloc((size_t)BHN * NQ * 4);
  float* gram      = (float*)alloc((size_t)BHN * DPAD * DPAD * 4);
  float* vnorm     = (float*)alloc((size_t)BHN * 4);
  int* samples     = (int*)alloc((size_t)BHN * SS * 4);
  __bf16* KpiBF    = (__bf16*)alloc((size_t)BHN * SS * DQ * 2);
  __bf16* WpiBF    = (__bf16*)alloc((size_t)BHN * SS * DPAD * 2);
  __bf16* qbf      = (__bf16*)alloc((size_t)BHN * NQ * DQ * 2);
  __bf16* kbf      = (__bf16*)alloc((size_t)BHN * NQ * DQ * 2);
  __bf16* wbf      = (__bf16*)alloc((size_t)BHN * NQ * DPAD * 2);
  float* res       = (float*)alloc((size_t)BHN * NQ * DP * 4);
  (void)ws_size; (void)in_sizes; (void)n_in; (void)out_size;

  k_hash      <<<BHN * NQ / 256, 256, 0, stream>>>(q, k, v, proj, codesQ, codesK, rownorm,
                                                   qbf, kbf, wbf);
  k_sort      <<<BHN * 2, 1024, 0, stream>>>(codesK, codesQ, Kidx, Qidx, invQ);
  k_block_attn<<<BHN * NB, 128, 0, stream>>>(qbf, kbf, wbf, Qidx, Kidx, res);
  k_gram      <<<BHN, 256, 0, stream>>>(wbf, gram);
  k_power     <<<BHN, 128, 0, stream>>>(gram, vnorm);
  k_sample    <<<BHN, 256, 0, stream>>>(rownorm, vnorm, Kidx, k, v, samples, KpiBF, WpiBF);
  k_residual  <<<BHN * (NQ / 64), 128, 0, stream>>>(qbf, invQ, samples, KpiBF, WpiBF, res);
  k_final     <<<BHN * NQ / 256, 256, 0, stream>>>(res, Qidx, out);
}
